// MolGNN_31688268710453
// MI455X (gfx1250) — matrix operand order repaired and checked by reading, not compiled, not run on hardware
//
#include <hip/hip_runtime.h>
#include <hip/hip_bf16.h>

typedef __attribute__((ext_vector_type(16))) __bf16 v16bf;
typedef __attribute__((ext_vector_type(8)))  float  v8f;
typedef unsigned short ushort_t;

typedef unsigned int tdm_u32x4 __attribute__((ext_vector_type(4)));
typedef int          tdm_i32x4 __attribute__((ext_vector_type(4)));
typedef int          tdm_i32x8 __attribute__((ext_vector_type(8)));

#define H 256
#define TM 128   // block tile M
#define TN 64    // block tile N
#define TK 32    // K chunk per WMMA

#define HAVE_TDM 0
#if defined(__has_builtin)
#if __has_builtin(__builtin_amdgcn_tensor_load_to_lds) && \
    __has_builtin(__builtin_amdgcn_s_wait_tensorcnt)
#undef HAVE_TDM
#define HAVE_TDM 1
#endif
#endif

__device__ __forceinline__ unsigned short f2bf(float f) {
    unsigned u = __float_as_uint(f);
    unsigned r = u + 0x7FFFu + ((u >> 16) & 1u);   // round-to-nearest-even
    return (unsigned short)(r >> 16);
}

union FragBF {
    uint4 q[2];
    v16bf v;
};

#if HAVE_TDM
// Issue one TDM load of a [TM x TK] bf16 tile from a row-major [M,256] tensor
// into LDS at lds_off.  Hardware zero-fills rows past rows_rem (OOB) and
// inserts 16B of LDS padding after every 64B row (pad_interval=16dw,
// pad_amount=4dw) -> LDS row stride = 80B = (TK+8) ushorts.
__device__ __forceinline__ void tdm_load_a_tile(unsigned lds_off,
                                                const ushort_t* gsrc,
                                                int rows_rem) {
    unsigned long long ga = (unsigned long long)(size_t)gsrc;
    unsigned rem = (unsigned)(rows_rem > 0 ? rows_rem : 0);
    tdm_u32x4 g0;
    g0[0] = 1u;                                            // count=1 (valid)
    g0[1] = lds_off;                                       // lds_addr
    g0[2] = (unsigned)ga;                                  // global_addr lo
    g0[3] = (unsigned)((ga >> 32) & 0x01FFFFFFull)         // global_addr hi
          | (2u << 30);                                    // type=2 (image)
    tdm_i32x8 g1;
    g1[0] = (int)((1u << 16)       // data_size = 2 bytes
                | (1u << 20)       // pad_enable
                | (3u << 22)       // pad_interval: 16 DWORDs (64B)
                | (3u << 25));     // pad_amount:   4 DWORDs (16B)
    g1[1] = (int)(((unsigned)H & 0xFFFFu) << 16);          // tensor_dim0 lo
    g1[2] = (int)(((rem & 0xFFFFu) << 16)                  // tensor_dim1 lo
                | (((unsigned)H >> 16) & 0xFFFFu));        // tensor_dim0 hi
    g1[3] = (int)(((unsigned)TK << 16)                     // tile_dim0 = 32
                | ((rem >> 16) & 0xFFFFu));                // tensor_dim1 hi
    g1[4] = (int)TM;                                       // tile_dim1 = 128
    g1[5] = (int)H;                                        // dim0_stride = 256
    g1[6] = 0;
    g1[7] = 0;
    tdm_i32x4 z4 = {0, 0, 0, 0};
#if __clang_major__ >= 23
    tdm_i32x8 z8 = {0, 0, 0, 0, 0, 0, 0, 0};
    __builtin_amdgcn_tensor_load_to_lds(g0, g1, z4, z4, z8, 0);
#else
    __builtin_amdgcn_tensor_load_to_lds(g0, g1, z4, z4, 0);
#endif
}
#endif

// ---------------------------------------------------------------------------
// Embedding sum kernels
// ---------------------------------------------------------------------------
__global__ void node_embed_kernel(const int* __restrict__ x,
                                  const float* __restrict__ tab,  // [9,119,256]
                                  float* __restrict__ h) {
    int n = blockIdx.x, c = threadIdx.x;
    float s = 0.f;
#pragma unroll
    for (int f = 0; f < 9; ++f) {
        int id = x[n * 9 + f];
        s += tab[((size_t)(f * 119 + id)) * H + c];
    }
    h[(size_t)n * H + c] = s;
}

__global__ void edge_embed_kernel(const int* __restrict__ ea,
                                  const float* __restrict__ tab,  // [3,22,256]
                                  float* __restrict__ e) {
    int ei = blockIdx.x, c = threadIdx.x;
    float s = 0.f;
#pragma unroll
    for (int f = 0; f < 3; ++f) {
        int id = ea[ei * 3 + f];
        s += tab[((size_t)(f * 22 + id)) * H + c];
    }
    e[(size_t)ei * H + c] = s;
}

// ---------------------------------------------------------------------------
// Utility kernels
// ---------------------------------------------------------------------------
__global__ void zero_f32_kernel(float* __restrict__ p, long long n4) {
    long long i = (long long)blockIdx.x * blockDim.x + threadIdx.x;
    long long stride = (long long)gridDim.x * blockDim.x;
    float4 z = make_float4(0.f, 0.f, 0.f, 0.f);
    for (; i < n4; i += stride) ((float4*)p)[i] = z;
}

// dst_bf16[i] = bf16(a[i] + (b ? b[i] : 0))
__global__ void pack_bf16_kernel(ushort_t* __restrict__ dst,
                                 const float* __restrict__ a,
                                 const float* __restrict__ b,
                                 long long n) {
    long long i = (long long)blockIdx.x * blockDim.x + threadIdx.x;
    long long stride = (long long)gridDim.x * blockDim.x;
    for (; i < n; i += stride) {
        float v = a[i];
        if (b) v += b[i];
        dst[i] = f2bf(v);
    }
}

// ---------------------------------------------------------------------------
// GINE message + scatter-add:  agg[dst] += relu(h[src] + e)
// ---------------------------------------------------------------------------
__global__ void msg_scatter_kernel(const float* __restrict__ h,
                                   const float* __restrict__ e,
                                   const int* __restrict__ src,
                                   const int* __restrict__ dst,
                                   float* __restrict__ agg) {
    int ei = blockIdx.x, c = threadIdx.x;
    int s = src[ei], d = dst[ei];
    float v = h[(size_t)s * H + c] + e[(size_t)ei * H + c];
    v = fmaxf(v, 0.f);
    atomicAdd(&agg[(size_t)d * H + c], v);
}

// global_add_pool:  gpool[batch[n]] += h[n]
__global__ void pool_scatter_kernel(const float* __restrict__ h,
                                    const int* __restrict__ batch,
                                    float* __restrict__ gpool) {
    int n = blockIdx.x, c = threadIdx.x;
    int g = batch[n];
    atomicAdd(&gpool[(size_t)g * H + c], h[(size_t)n * H + c]);
}

// ---------------------------------------------------------------------------
// bf16 WMMA GEMM:  out[M,Nout] = act( A[M,256] @ W[256,Nout] + bias )
// A, W bf16 row-major, K fixed 256.  Block 256 threads (8 waves, 4x2),
// block tile 128x64, wave tile 32x32 (2x2 WMMA tiles).
// A streamed via TDM tensor_load_to_lds, double buffered (overlap w/ WMMA).
// ---------------------------------------------------------------------------
__global__ __launch_bounds__(256)
void gemm_bf16_kernel(const ushort_t* __restrict__ A,
                      const ushort_t* __restrict__ W,
                      const float* __restrict__ bias,
                      void* __restrict__ out,
                      int M, int Nout, int relu, int out_bf16) {
    __shared__ ushort_t Bs[TN][H + 8];          // W tile, transposed: Bs[col][k]
    __shared__ ushort_t As[2][TM][TK + 8];      // A chunks, double buffered

    const int tid = threadIdx.x;
    const int w   = tid >> 5;            // wave id, wave32
    const int t   = tid & 31;            // lane
    const int wm  = w >> 1;              // 0..3
    const int wn  = w & 1;               // 0..1
    const int m0  = blockIdx.x * TM;
    const int n0  = blockIdx.y * TN;
    const int lane16 = t & 15;
    const bool hi    = t >= 16;

    // --- load full-K W tile (transposed) into LDS: 64 cols x 256 k ---
    for (int i = tid; i < (H * TN) / 8; i += 256) {
        int k  = i >> 3;
        int c8 = (i & 7) * 8;
        uint4 val = *(const uint4*)&W[(size_t)k * Nout + n0 + c8];
        const ushort_t* p = (const ushort_t*)&val;
#pragma unroll
        for (int j = 0; j < 8; ++j) Bs[c8 + j][k] = p[j];
    }

#if HAVE_TDM
    const unsigned as_base = (unsigned)(size_t)(void*)&As[0][0][0];
    const unsigned as_sz   = (unsigned)(sizeof(ushort_t) * TM * (TK + 8));
    const ushort_t* arow   = A + (size_t)m0 * H;
    const int rem          = M - m0;
    if (tid == 0) {
        tdm_load_a_tile(as_base, arow, rem);               // prefetch buf 0
        __builtin_amdgcn_s_wait_tensorcnt(0);
    }
#endif

    v8f acc[2][2];
#pragma unroll
    for (int i = 0; i < 2; ++i)
#pragma unroll
        for (int j = 0; j < 2; ++j) acc[i][j] = (v8f)(0.f);

    int buf = 0;
    __syncthreads();   // W tile ready (+ first A tile under TDM)

    for (int kc = 0; kc < H; kc += TK) {
#if HAVE_TDM
        if ((kc + TK < H) && tid == 0)
            tdm_load_a_tile(as_base + (unsigned)(buf ^ 1) * as_sz,
                            arow + kc + TK, rem);
#else
        // fallback: cooperative load of A chunk [128 x 32] into As[0]
        for (int i = tid; i < (TM * TK) / 8; i += 256) {
            int r  = i >> 2;
            int sg = (i & 3) * 8;
            int gr = m0 + r;
            uint4 v = make_uint4(0u, 0u, 0u, 0u);
            if (gr < M) v = *(const uint4*)&A[(size_t)gr * H + kc + sg];
            *(uint4*)&As[0][r][sg] = v;
        }
        __syncthreads();
#endif

        // --- build fragments per ISA 7.12.2 layouts ---
        FragBF aF[2], bF[2];
        {
            int kb = hi ? 8 : 0;   // A: lanes16-31 hold K 8-15 / 24-31
#pragma unroll
            for (int i = 0; i < 2; ++i) {
                int r = wm * 32 + i * 16 + lane16;
                aF[i].q[0] = *(const uint4*)&As[buf][r][kb];
                aF[i].q[1] = *(const uint4*)&As[buf][r][kb + 16];
            }
            int kb2 = hi ? 8 : 0;  // B: lanes16-31 hold K 8-15 / 24-31
#pragma unroll
            for (int j = 0; j < 2; ++j) {
                int c = wn * 32 + j * 16 + lane16;
                bF[j].q[0] = *(const uint4*)&Bs[c][kc + kb2];
                bF[j].q[1] = *(const uint4*)&Bs[c][kc + kb2 + 16];
            }
        }
#pragma unroll
        for (int i = 0; i < 2; ++i)
#pragma unroll
            for (int j = 0; j < 2; ++j)
                acc[i][j] = __builtin_amdgcn_wmma_f32_16x16x32_bf16(
                    false, aF[i].v, false, bF[j].v,
                    (short)0, acc[i][j], false, false);

#if HAVE_TDM
        if (tid == 0) __builtin_amdgcn_s_wait_tensorcnt(0);
        __syncthreads();
        buf ^= 1;
#else
        __syncthreads();
#endif
    }

    // --- epilogue: bias + relu + store (C layout: col=lane16, rows v / v+8) ---
    const int rb = hi ? 8 : 0;
    float* outf = (float*)out;
    ushort_t* outb = (ushort_t*)out;
#pragma unroll
    for (int i = 0; i < 2; ++i) {
#pragma unroll
        for (int j = 0; j < 2; ++j) {
            int col = n0 + wn * 32 + j * 16 + lane16;
            float bv = bias ? bias[col] : 0.f;
#pragma unroll
            for (int v = 0; v < 8; ++v) {
                int row = m0 + wm * 32 + i * 16 + rb + v;
                if (row < M) {
                    float val = acc[i][j][v] + bv;
                    if (relu) val = fmaxf(val, 0.f);
                    if (out_bf16) outb[(size_t)row * Nout + col] = f2bf(val);
                    else          outf[(size_t)row * Nout + col] = val;
                }
            }
        }
    }
}

// ---------------------------------------------------------------------------
// Row-wise L2 normalize: g /= max(||g||, 1e-12).  One block per row.
// ---------------------------------------------------------------------------
__global__ void l2norm_kernel(float* __restrict__ g, int cols) {
    __shared__ float red[8];
    __shared__ float inv_s;
    int row = blockIdx.x, tid = threadIdx.x;
    float s = 0.f;
    for (int c = tid; c < cols; c += blockDim.x) {
        float v = g[(size_t)row * cols + c];
        s += v * v;
    }
#pragma unroll
    for (int off = 16; off > 0; off >>= 1) s += __shfl_xor(s, off, 32);
    if ((tid & 31) == 0) red[tid >> 5] = s;
    __syncthreads();
    if (tid == 0) {
        float tot = 0.f;
#pragma unroll
        for (int i = 0; i < 8; ++i) tot += red[i];
        inv_s = 1.f / fmaxf(sqrtf(tot), 1e-12f);
    }
    __syncthreads();
    float inv = inv_s;
    for (int c = tid; c < cols; c += blockDim.x)
        g[(size_t)row * cols + c] *= inv;
}

// ---------------------------------------------------------------------------
// Host launch
// ---------------------------------------------------------------------------
extern "C" void kernel_launch(void* const* d_in, const int* in_sizes, int n_in,
                              void* d_out, int out_size, void* d_ws, size_t ws_size,
                              hipStream_t stream) {
    const int* x        = (const int*)d_in[0];
    const int* eidx     = (const int*)d_in[1];
    const int* eattr    = (const int*)d_in[2];
    const int* batch    = (const int*)d_in[3];
    const float* node_tab = (const float*)d_in[4];
    const float* edge_tab = (const float*)d_in[5];
    const float* ep_W1  = (const float*)d_in[6];
    const float* ep_b1  = (const float*)d_in[7];
    const float* ep_W2  = (const float*)d_in[8];
    const float* ep_b2  = (const float*)d_in[9];
    const float* conv_W1 = (const float*)d_in[10];
    const float* conv_b1 = (const float*)d_in[11];
    const float* conv_W2 = (const float*)d_in[12];
    const float* conv_b2 = (const float*)d_in[13];
    const float* proj_W = (const float*)d_in[14];
    const float* proj_b = (const float*)d_in[15];

    const int N = in_sizes[3];
    const int E = in_sizes[1] / 2;
    const int OUT = 512;
    const int G = out_size / OUT;
    const int L = in_sizes[10] / (H * H);

    const int* src = eidx;        // edge_index[0]
    const int* dst = eidx + E;    // edge_index[1]

    // workspace carve-out
    char* ws = (char*)d_ws;
    size_t off = 0;
    auto take = [&](size_t bytes) -> void* {
        void* p = ws + off;
        off += (bytes + 255) & ~(size_t)255;
        return p;
    };
    float* h     = (float*)take((size_t)N * H * 4);
    float* e     = (float*)take((size_t)E * H * 4);
    float* agg   = (float*)take((size_t)N * H * 4);
    float* gpool = (float*)take((size_t)G * H * 4);
    ushort_t* abf  = (ushort_t*)take((size_t)E * H * 2);
    ushort_t* abf2 = (ushort_t*)take((size_t)E * H * 2);
    ushort_t* wb   = (ushort_t*)take((size_t)(H * H * (2 + 2 * L) + H * OUT) * 2);

    ushort_t* wb_epW1 = wb;
    ushort_t* wb_epW2 = wb_epW1 + H * H;
    ushort_t* wb_c1   = wb_epW2 + H * H;        // L * H*H
    ushort_t* wb_c2   = wb_c1 + (size_t)L * H * H;
    ushort_t* wb_proj = wb_c2 + (size_t)L * H * H;

    // 1. weights -> bf16
    pack_bf16_kernel<<<256, 256, 0, stream>>>(wb_epW1, ep_W1, nullptr, (long long)H * H);
    pack_bf16_kernel<<<256, 256, 0, stream>>>(wb_epW2, ep_W2, nullptr, (long long)H * H);
    pack_bf16_kernel<<<512, 256, 0, stream>>>(wb_c1, conv_W1, nullptr, (long long)L * H * H);
    pack_bf16_kernel<<<512, 256, 0, stream>>>(wb_c2, conv_W2, nullptr, (long long)L * H * H);
    pack_bf16_kernel<<<256, 256, 0, stream>>>(wb_proj, proj_W, nullptr, (long long)H * OUT);

    // 2. embeddings
    node_embed_kernel<<<N, H, 0, stream>>>(x, node_tab, h);
    edge_embed_kernel<<<E, H, 0, stream>>>(eattr, edge_tab, e);

    // 3. edge MLP (bf16 WMMA): e = relu(e@W1+b1)@W2+b2
    pack_bf16_kernel<<<2048, 256, 0, stream>>>(abf, e, nullptr, (long long)E * H);
    dim3 gE((E + TM - 1) / TM, H / TN);
    gemm_bf16_kernel<<<gE, 256, 0, stream>>>(abf, wb_epW1, ep_b1, abf2, E, H, 1, 1);
    gemm_bf16_kernel<<<gE, 256, 0, stream>>>(abf2, wb_epW2, ep_b2, e, E, H, 0, 0);

    // 4. GINE layers
    dim3 gN((N + TM - 1) / TM, H / TN);
    for (int l = 0; l < L; ++l) {
        zero_f32_kernel<<<2048, 256, 0, stream>>>(agg, (long long)N * H / 4);
        msg_scatter_kernel<<<E, H, 0, stream>>>(h, e, src, dst, agg);
        pack_bf16_kernel<<<2048, 256, 0, stream>>>(abf, h, agg, (long long)N * H);
        gemm_bf16_kernel<<<gN, 256, 0, stream>>>(abf, wb_c1 + (size_t)l * H * H,
                                                 conv_b1 + l * H, abf2, N, H, 1, 1);
        gemm_bf16_kernel<<<gN, 256, 0, stream>>>(abf2, wb_c2 + (size_t)l * H * H,
                                                 conv_b2 + l * H, h, N, H, 1, 0);
    }

    // 5. pool + projection + normalize
    zero_f32_kernel<<<1024, 256, 0, stream>>>(gpool, (long long)G * H / 4);
    pool_scatter_kernel<<<N, H, 0, stream>>>(h, batch, gpool);
    pack_bf16_kernel<<<1024, 256, 0, stream>>>(abf, gpool, nullptr, (long long)G * H);
    dim3 gP((G + TM - 1) / TM, OUT / TN);
    gemm_bf16_kernel<<<gP, 256, 0, stream>>>(abf, wb_proj, proj_b, d_out, G, OUT, 0, 0);
    l2norm_kernel<<<G, 256, 0, stream>>>((float*)d_out, OUT);
}
